// LatentNeuralODE_68762426409339
// MI455X (gfx1250) — matrix operand order repaired and checked by reading, not compiled, not run on hardware
//
#include <hip/hip_runtime.h>

typedef __attribute__((ext_vector_type(16))) _Float16 v16h;
typedef __attribute__((ext_vector_type(8)))  _Float16 v8h;
typedef __attribute__((ext_vector_type(8)))  float    v8f;
typedef __attribute__((ext_vector_type(2)))  float    v2f;

#define NW   8     // waves per block (wave32)
#define HDIM 64

// ---------------- WMMA wrappers ----------------
__device__ __forceinline__ v8f wmma_f16(v16h a, v16h b, v8f c) {
  return __builtin_amdgcn_wmma_f32_16x16x32_f16(false, a, false, b, (short)0, c, false, false);
}
__device__ __forceinline__ v8f wmma_f32k4(v2f a, v2f b, v8f c) {
  return __builtin_amdgcn_wmma_f32_16x16x4_f32(false, a, false, b, (short)0, c, false, false);
}

// ---------------- fast transcendentals (trans pipe, no IEEE div chains) ----------------
__device__ __forceinline__ float fast_rcp(float x) {
#if __has_builtin(__builtin_amdgcn_rcpf)
  return __builtin_amdgcn_rcpf(x);           // v_rcp_f32
#else
  return 1.f / x;
#endif
}
__device__ __forceinline__ float fast_tanh(float x) {
#if __has_builtin(__builtin_amdgcn_tanhf)
  return __builtin_amdgcn_tanhf(x);          // native v_tanh_f32 (gfx1250 trans op)
#elif __has_builtin(__builtin_amdgcn_tanh_f32)
  return __builtin_amdgcn_tanh_f32(x);
#else
  // tanh(x) = 1 - 2/(e^{2x}+1): one v_exp_f32 + one v_rcp_f32
  const float e = __expf(2.f * x);
  return 1.f - 2.f * fast_rcp(e + 1.f);
#endif
}
__device__ __forceinline__ float fast_sigmoid(float x) {
  return fast_rcp(1.f + __expf(-x));         // v_exp_f32 + v_rcp_f32
}

// A operand (16x32 f16 tile) from row-major f16 matrix in LDS.
// lane<16: K = kb+0..15; lane>=16: K = kb+16..31
__device__ __forceinline__ v16h load_a_lds(const _Float16* W, int ld, int row0, int kbase, int lane) {
  const int r  = row0 + (lane & 15);
  const int k0 = kbase + ((lane & 16) ? 16 : 0);
  v8h lo = *(const v8h*)(W + r * ld + k0);
  v8h hi = *(const v8h*)(W + r * ld + k0 + 8);
  v16h a;
#pragma unroll
  for (int j = 0; j < 8; ++j) { a[j] = lo[j]; a[8 + j] = hi[j]; }
  return a;
}

// A operand (16x32 f16 tile) from row-major f32 matrix in global; zero-pads rows >= nrows.
__device__ __forceinline__ v16h load_a_gl(const float* W, int ld, int nrows, int row0, int kbase, int lane) {
  const int r  = row0 + (lane & 15);
  const int k0 = kbase + ((lane & 16) ? 16 : 0);
  v16h a;
  if (r < nrows) {
    const float* p = W + r * ld;
#pragma unroll
    for (int j = 0; j < 8; ++j) { a[j] = (_Float16)p[k0 + j]; a[8 + j] = (_Float16)p[k0 + 8 + j]; }
  } else {
#pragma unroll
    for (int j = 0; j < 16; ++j) a[j] = (_Float16)0.f;
  }
  return a;
}

// B operand (32x16 f16 chunk) from per-wave LDS activation buffer h[batch16][HDIM].
// lane<16: K = kbase+0..15 ; lane>=16: K = kbase+16..31 (column = lane%16)
__device__ __forceinline__ v16h load_b16(const _Float16* hbuf, int kbase, int lane) {
  const int c  = lane & 15;
  const int k0 = kbase + ((lane & 16) ? 16 : 0);
  return *(const v16h*)(hbuf + c * HDIM + k0);
}

// D (16x16 f32, rows=features, cols=batch) -> tanh -> f16 into h[batch16][HDIM]
__device__ __forceinline__ void store_d_tanh(_Float16* hbuf, v8f d, int mtile, int lane) {
  const int c  = lane & 15;
  const int r0 = mtile * 16 + ((lane & 16) ? 8 : 0);
  v8h hh;
#pragma unroll
  for (int j = 0; j < 8; ++j) hh[j] = (_Float16)fast_tanh(d[j]);
  *(v8h*)(hbuf + c * HDIM + r0) = hh;
}

// ================= Kernel 1: GRU encoder + y0 = softmax(Linear(hT)) =================
__global__ void __launch_bounds__(32 * NW)
gru_encode_kernel(const float* __restrict__ x,
                  const float* __restrict__ W_ih, const float* __restrict__ W_hh,
                  const float* __restrict__ b_ih, const float* __restrict__ b_hh,
                  const float* __restrict__ W_y0, const float* __restrict__ b_y0,
                  const int* __restrict__ Tptr,
                  float* __restrict__ y0_ws, float* __restrict__ out, int B) {
  const int  lane = threadIdx.x & 31;
  const int  w    = threadIdx.x >> 5;
  const int  col  = lane & 15;
  const bool blo  = lane < 16;
  const int  m0   = blo ? 0 : 8;
  const int  b    = (blockIdx.x * NW + w) * 16 + col;
  const bool ok   = b < B;
  const int  T    = *Tptr;

  __shared__ _Float16 sWhh[192 * HDIM];     // 24 KB, A-source for gate GEMMs
  __shared__ float    sWih[192 * 3];
  __shared__ float    sbih[192];
  __shared__ float    sbhh[192];
  __shared__ float    sWy0[3 * HDIM];
  __shared__ float    sby0[3];
  __shared__ _Float16 sh[NW][16 * HDIM];    // per-wave h state, f16, [batch][feat]
  __shared__ float    sx[NW][16 * 3];

  for (int i = threadIdx.x; i < 192 * HDIM; i += blockDim.x) sWhh[i] = (_Float16)W_hh[i];
  for (int i = threadIdx.x; i < 192 * 3; i += blockDim.x) sWih[i] = W_ih[i];
  for (int i = threadIdx.x; i < 192; i += blockDim.x) { sbih[i] = b_ih[i]; sbhh[i] = b_hh[i]; }
  for (int i = threadIdx.x; i < 3 * HDIM; i += blockDim.x) sWy0[i] = W_y0[i];
  if (threadIdx.x < 3) sby0[threadIdx.x] = b_y0[threadIdx.x];
  for (int i = lane; i < 16 * HDIM; i += 32) sh[w][i] = (_Float16)0.f;  // h0 = 0
  __syncthreads();

  for (int s = 0; s < 10; ++s) {
    if (blo) {
      const float* xp = x + ((long)(ok ? b : 0) * 10 + s) * 3;
      sx[w][col * 3 + 0] = ok ? xp[0] : 0.f;
      sx[w][col * 3 + 1] = ok ? xp[1] : 0.f;
      sx[w][col * 3 + 2] = ok ? xp[2] : 0.f;
    }
    // B operand = h_prev^T (K = 64 in two chunks)
    v16h bh0 = load_b16(sh[w], 0, lane);
    v16h bh1 = load_b16(sh[w], 32, lane);
    const float xv0 = sx[w][col * 3 + 0];
    const float xv1 = sx[w][col * 3 + 1];
    const float xv2 = sx[w][col * 3 + 2];

    for (int f = 0; f < 4; ++f) {               // 16-feature slab of h
      v8h ph = *(const v8h*)(sh[w] + col * HDIM + f * 16 + m0);  // h_prev slots
      v8f dg[3];                                 // gh for r,z,n (b_hh folded via C)
#pragma unroll
      for (int g = 0; g < 3; ++g) {
        const int mt = g * 4 + f;
        v8f c;
#pragma unroll
        for (int j = 0; j < 8; ++j) c[j] = sbhh[mt * 16 + m0 + j];
        v16h a0 = load_a_lds(sWhh, HDIM, mt * 16, 0, lane);
        v16h a1 = load_a_lds(sWhh, HDIM, mt * 16, 32, lane);
        v8f  d  = wmma_f16(a0, bh0, c);
        dg[g]   = wmma_f16(a1, bh1, d);
      }
      v8h hnew;
#pragma unroll
      for (int j = 0; j < 8; ++j) {
        const int F  = f * 16 + m0 + j;
        const int Fr = F, Fz = 64 + F, Fn = 128 + F;
        float gir = sbih[Fr] + xv0 * sWih[Fr * 3] + xv1 * sWih[Fr * 3 + 1] + xv2 * sWih[Fr * 3 + 2];
        float giz = sbih[Fz] + xv0 * sWih[Fz * 3] + xv1 * sWih[Fz * 3 + 1] + xv2 * sWih[Fz * 3 + 2];
        float gin = sbih[Fn] + xv0 * sWih[Fn * 3] + xv1 * sWih[Fn * 3 + 1] + xv2 * sWih[Fn * 3 + 2];
        float r = fast_sigmoid(gir + dg[0][j]);
        float z = fast_sigmoid(giz + dg[1][j]);
        float n = fast_tanh(gin + r * dg[2][j]);
        hnew[j] = (_Float16)((1.f - z) * n + z * (float)ph[j]);
      }
      *(v8h*)(sh[w] + col * HDIM + f * 16 + m0) = hnew;
    }
  }

  // y0 = softmax(hT @ W_y0^T + b_y0), store to ws and out[:,0,:]
  if (blo) {
    float l0 = sby0[0], l1 = sby0[1], l2 = sby0[2];
    for (int fd = 0; fd < HDIM; ++fd) {
      float hv = (float)sh[w][col * HDIM + fd];
      l0 += hv * sWy0[0 * HDIM + fd];
      l1 += hv * sWy0[1 * HDIM + fd];
      l2 += hv * sWy0[2 * HDIM + fd];
    }
    float mx = fmaxf(l0, fmaxf(l1, l2));
    float e0 = __expf(l0 - mx), e1 = __expf(l1 - mx), e2 = __expf(l2 - mx);
    float inv = fast_rcp(e0 + e1 + e2);
    if (ok) {
      y0_ws[b * 3 + 0] = e0 * inv;
      y0_ws[b * 3 + 1] = e1 * inv;
      y0_ws[b * 3 + 2] = e2 * inv;
      out[((long)b * T + 0) * 3 + 0] = e0 * inv;
      out[((long)b * T + 0) * 3 + 1] = e1 * inv;
      out[((long)b * T + 0) * 3 + 2] = e2 * inv;
    }
  }
}

// ================= Kernel 2: dopri5 Neural-ODE integration =================
__global__ void __launch_bounds__(32 * NW)
ode_kernel(const float* __restrict__ W1, const float* __restrict__ b1,
           const float* __restrict__ W2, const float* __restrict__ b2,
           const float* __restrict__ W3, const float* __restrict__ b3,
           const int* __restrict__ Tptr, const float* __restrict__ y0_ws,
           float* __restrict__ out, int B) {
  const int  lane = threadIdx.x & 31;
  const int  w    = threadIdx.x >> 5;
  const int  col  = lane & 15;
  const bool blo  = lane < 16;
  const int  m0   = blo ? 0 : 8;
  const int  b    = (blockIdx.x * NW + w) * 16 + col;
  const bool ok   = b < B;
  const int  T    = *Tptr;

  __shared__ _Float16 sh[NW][16 * HDIM];
  _Float16* shw = sh[w];

  // Layer1 (fp32 WMMA, K=4): A = [W1 | b1], B supplies [y ; 1] => D = W1*y + b1
  v2f a1[4];
#pragma unroll
  for (int m = 0; m < 4; ++m) {
    const int r = m * 16 + col;
    if (blo) { a1[m][0] = W1[r * 3 + 0]; a1[m][1] = W1[r * 3 + 1]; }
    else     { a1[m][0] = W1[r * 3 + 2]; a1[m][1] = b1[r]; }
  }
  // Layer2 (f16 WMMA, K=64 in 2 chunks), bias b2 folded into C
  v16h a2[4][2];
  v8f  c2[4];
#pragma unroll
  for (int m = 0; m < 4; ++m) {
    a2[m][0] = load_a_gl(W2, HDIM, HDIM, m * 16, 0, lane);
    a2[m][1] = load_a_gl(W2, HDIM, HDIM, m * 16, 32, lane);
#pragma unroll
    for (int j = 0; j < 8; ++j) c2[m][j] = b2[m * 16 + m0 + j];
  }
  // Layer3: W3 (2x64) padded to 16 rows, b3 via C rows 0..1
  v16h a3[2];
  a3[0] = load_a_gl(W3, HDIM, 2, 0, 0, lane);
  a3[1] = load_a_gl(W3, HDIM, 2, 0, 32, lane);
  v8f c3 = {};
  if (blo) { c3[0] = b3[0]; c3[1] = b3[1]; }

  float y[3] = {0.f, 0.f, 0.f};
  if (blo && ok) { y[0] = y0_ws[b * 3]; y[1] = y0_ws[b * 3 + 1]; y[2] = y0_ws[b * 3 + 2]; }

  auto feval = [&](const float* yv, float* kv) {
    // B for layer1: lane's column owner is lane%16 (low lanes hold y)
    const float q0 = __shfl(yv[0], col, 32);
    const float q1 = __shfl(yv[1], col, 32);
    const float q2 = __shfl(yv[2], col, 32);
    v2f bv;
    if (blo) { bv[0] = q0; bv[1] = q1; }
    else     { bv[0] = q2; bv[1] = 1.0f; }  // K=3 -> y2, K=4 slot -> 1 (bias lane)
    const v8f z8 = {};
#pragma unroll
    for (int m = 0; m < 4; ++m) {           // h1 = tanh(W1*y + b1)
      v8f d = wmma_f32k4(a1[m], bv, z8);
      store_d_tanh(shw, d, m, lane);
    }
    v16h bb0 = load_b16(shw, 0, lane);
    v16h bb1 = load_b16(shw, 32, lane);
#pragma unroll
    for (int m = 0; m < 4; ++m) {           // h2 = tanh(W2*h1 + b2)
      v8f d = wmma_f16(a2[m][0], bb0, c2[m]);
      d     = wmma_f16(a2[m][1], bb1, d);
      store_d_tanh(shw, d, m, lane);
    }
    bb0 = load_b16(shw, 0, lane);
    bb1 = load_b16(shw, 32, lane);
    v8f d3 = wmma_f16(a3[0], bb0, c3);      // out = W3*h2 + b3
    d3     = wmma_f16(a3[1], bb1, d3);
    const float o0 = d3[0], o1 = d3[1];     // lanes<16: rows 0,1 = out0,out1
    const float dS = -fmaxf(o0, 0.f) * fmaxf(yv[0], 0.f);
    const float dR =  fmaxf(o1, 0.f) * fmaxf(yv[1], 0.f);
    kv[0] = dS; kv[1] = -(dS + dR); kv[2] = dR;
  };

  const float hs = 0.25f;  // 1/NSUB
  for (int t = 1; t < T; ++t) {
    for (int sub = 0; sub < 4; ++sub) {
      float k1[3], k2[3], k3[3], k4[3], k5[3], k6[3], yt[3];
      feval(y, k1);
#pragma unroll
      for (int c = 0; c < 3; ++c) yt[c] = y[c] + hs * (0.2f * k1[c]);
      feval(yt, k2);
#pragma unroll
      for (int c = 0; c < 3; ++c) yt[c] = y[c] + hs * ((3.f/40.f)*k1[c] + (9.f/40.f)*k2[c]);
      feval(yt, k3);
#pragma unroll
      for (int c = 0; c < 3; ++c) yt[c] = y[c] + hs * ((44.f/45.f)*k1[c] - (56.f/15.f)*k2[c] + (32.f/9.f)*k3[c]);
      feval(yt, k4);
#pragma unroll
      for (int c = 0; c < 3; ++c) yt[c] = y[c] + hs * ((19372.f/6561.f)*k1[c] - (25360.f/2187.f)*k2[c]
                                                      + (64448.f/6561.f)*k3[c] - (212.f/729.f)*k4[c]);
      feval(yt, k5);
#pragma unroll
      for (int c = 0; c < 3; ++c) yt[c] = y[c] + hs * ((9017.f/3168.f)*k1[c] - (355.f/33.f)*k2[c]
                                                      + (46732.f/5247.f)*k3[c] + (49.f/176.f)*k4[c]
                                                      - (5103.f/18656.f)*k5[c]);
      feval(yt, k6);
#pragma unroll
      for (int c = 0; c < 3; ++c) y[c] += hs * ((35.f/384.f)*k1[c] + (500.f/1113.f)*k3[c]
                                               + (125.f/192.f)*k4[c] - (2187.f/6784.f)*k5[c]
                                               + (11.f/84.f)*k6[c]);
    }
    if (blo && ok) {
      out[((long)b * T + t) * 3 + 0] = y[0];
      out[((long)b * T + t) * 3 + 1] = y[1];
      out[((long)b * T + t) * 3 + 2] = y[2];
    }
  }
}

// ================= host launcher =================
extern "C" void kernel_launch(void* const* d_in, const int* in_sizes, int n_in,
                              void* d_out, int out_size, void* d_ws, size_t ws_size,
                              hipStream_t stream) {
  (void)n_in; (void)out_size; (void)ws_size;
  const float* x    = (const float*)d_in[0];
  const float* Wih  = (const float*)d_in[1];
  const float* Whh  = (const float*)d_in[2];
  const float* bih  = (const float*)d_in[3];
  const float* bhh  = (const float*)d_in[4];
  const float* Wy0  = (const float*)d_in[5];
  const float* by0  = (const float*)d_in[6];
  const float* W1   = (const float*)d_in[7];
  const float* b1   = (const float*)d_in[8];
  const float* W2   = (const float*)d_in[9];
  const float* b2   = (const float*)d_in[10];
  const float* W3   = (const float*)d_in[11];
  const float* b3   = (const float*)d_in[12];
  const int*   Tptr = (const int*)d_in[13];

  const int B = in_sizes[0] / 30;  // x is (B, 10, 3)
  float* y0ws = (float*)d_ws;      // B*3 floats
  float* out  = (float*)d_out;

  const int tiles  = (B + 15) / 16;
  const int blocks = (tiles + NW - 1) / NW;

  gru_encode_kernel<<<blocks, 32 * NW, 0, stream>>>(x, Wih, Whh, bih, bhh, Wy0, by0,
                                                    Tptr, y0ws, out, B);
  ode_kernel<<<blocks, 32 * NW, 0, stream>>>(W1, b1, W2, b2, W3, b3,
                                             Tptr, y0ws, out, B);
}
